// ST_Enc_70385924047395
// MI455X (gfx1250) — compile-verified, matrix operand order checked by reading, not run on hardware
//
#include <hip/hip_runtime.h>
#include <hip/hip_bf16.h>
#include <math.h>

#define B_   16
#define TIN_ 12
#define N_   512
#define FIN_ 2
#define TP_  24
#define CR_  64
#define CC_  64
#define CS_  128
#define CE_  256

typedef __attribute__((ext_vector_type(16))) _Float16 v16h;
typedef __attribute__((ext_vector_type(8)))  float    v8f;
typedef __attribute__((ext_vector_type(4)))  unsigned int u32x4;
typedef __attribute__((ext_vector_type(8)))  unsigned int u32x8;

#define WMMA(a,b,c) __builtin_amdgcn_wmma_f32_16x16x32_f16(false,(a),false,(b),(short)0,(c),false,false)

// ---- Tensor Data Mover: stage a 2D f16 tile (global -> LDS), D# per ISA ch.8 ----
// group0: [1:0]=count=1, [63:32]=lds_addr, [120:64]=global_addr, [127:126]=type=2
// group1: [17:16]=data_size(1=2B), [79:48]=tensor_dim0, [111:80]=tensor_dim1,
//         [127:112]=tile_dim0, [143:128]=tile_dim1, [207:160]=tensor_dim0_stride
__device__ __forceinline__ void tdm_load_2d_f16(unsigned lds_off, const void* gptr,
                                                unsigned tensor_d0, unsigned tensor_d1,
                                                unsigned tile_d0, unsigned tile_d1,
                                                unsigned long long stride0) {
  unsigned long long ga = (unsigned long long)gptr;
  u32x4 g0;
  g0[0] = 1u;                                               // count=1, user mode
  g0[1] = lds_off;                                          // D#.lds_addr
  g0[2] = (unsigned)ga;                                     // global_addr[31:0]
  g0[3] = (unsigned)((ga >> 32) & 0x1FFFFFFull) | (2u << 30); // addr[56:32] | type=2
  u32x8 g1;
  g1[0] = (1u << 16);                                       // data_size = 2 bytes
  g1[1] = (tensor_d0 & 0xFFFFu) << 16;                      // tensor_dim0[15:0]
  g1[2] = (tensor_d0 >> 16) | ((tensor_d1 & 0xFFFFu) << 16);
  g1[3] = (tensor_d1 >> 16) | (tile_d0 << 16);              // tile_dim0
  g1[4] = tile_d1 & 0xFFFFu;                                // tile_dim1 (tile_dim2=0)
  g1[5] = (unsigned)stride0;                                // tensor_dim0_stride[31:0]
  g1[6] = (unsigned)(stride0 >> 32) & 0xFFFFu;              // stride0[47:32]
  g1[7] = 0u;
  asm volatile("tensor_load_to_lds %0, %1" :: "s"(g0), "s"(g1) : "memory");
}

// ---- WMMA fragment loaders (ISA 7.12.2 16-bit A 16x32 / B 32x16 striping) ----
// element (mn, k) = s[mn*strideMN + k*strideK]; VGPR j holds K pair
//   k0 = 16*(j>>2) + 2*(j&3) + (lane>=16 ? 8 : 0)
__device__ __forceinline__ v16h ld_frag_h(const _Float16* s, int strideMN, int strideK) {
  int lane = threadIdx.x & 31;
  int mn = lane & 15;
  int hi = (lane & 16) ? 8 : 0;
  v16h v;
#pragma unroll
  for (int j = 0; j < 8; ++j) {
    int k0 = ((j >> 2) << 4) + ((j & 3) << 1) + hi;
    v[2*j]   = s[(size_t)mn*strideMN + (size_t)k0*strideK];
    v[2*j+1] = s[(size_t)mn*strideMN + (size_t)(k0+1)*strideK];
  }
  return v;
}
__device__ __forceinline__ v16h ld_frag_f(const float* s, int strideMN, int strideK) {
  int lane = threadIdx.x & 31;
  int mn = lane & 15;
  int hi = (lane & 16) ? 8 : 0;
  v16h v;
#pragma unroll
  for (int j = 0; j < 8; ++j) {
    int k0 = ((j >> 2) << 4) + ((j & 3) << 1) + hi;
    v[2*j]   = (_Float16)s[(size_t)mn*strideMN + (size_t)k0*strideK];
    v[2*j+1] = (_Float16)s[(size_t)mn*strideMN + (size_t)(k0+1)*strideK];
  }
  return v;
}

// ---- small elementwise kernels ----
__global__ void k_f2h(const float* s, _Float16* d, int n) {
  int i = blockIdx.x*256 + threadIdx.x;
  if (i < n) d[i] = (_Float16)s[i];
}
__global__ void k_zerof(float* p, int n) {
  int i = blockIdx.x*256 + threadIdx.x;
  if (i < n) p[i] = 0.f;
}

// inp[b,f,n,p] = sum_t source[b,t,n,f]*plus_w[p,t] + plus_b[p]
__global__ void k_embed(const float* src, const float* pw, const float* pb, float* inp) {
  int i = blockIdx.x*256 + threadIdx.x;
  if (i >= B_*FIN_*N_*TP_) return;
  int p = i % TP_; int n = (i/TP_) % N_; int f = (i/(TP_*N_)) % FIN_; int b = i/(TP_*N_*FIN_);
  float s = pb[p];
  for (int t = 0; t < TIN_; ++t)
    s += src[(((size_t)b*TIN_ + t)*N_ + n)*FIN_ + f] * pw[p*TIN_ + t];
  inp[i] = s;
}

// xa[b,t,n,c] = ext 1x1 conv of inp (attention layout)
__global__ void k_xa(const float* inp, const float* ew, const float* eb, float* xa) {
  int i = blockIdx.x*256 + threadIdx.x;
  if (i >= B_*TP_*N_*CR_) return;
  int c = i % CR_; int n = (i/CR_) % N_; int t = (i/(CR_*N_)) % TP_; int b = i/(CR_*N_*TP_);
  float v = eb[c];
  v += inp[(((size_t)b*FIN_ + 0)*N_ + n)*TP_ + t] * ew[c*FIN_ + 0];
  v += inp[(((size_t)b*FIN_ + 1)*N_ + n)*TP_ + t] * ew[c*FIN_ + 1];
  xa[i] = v;
}

// qkv = xa @ xff_w^T + xff_b, split -> q,k,v (f16).
// Weight panel (192x64 f16, 24KB) staged into LDS via TDM; 128 rows per wave.
__global__ __launch_bounds__(32)
void k_qkv(const float* xa, const _Float16* wh, const float* bias,
           _Float16* qh, _Float16* kh, _Float16* vh) {
  __shared__ _Float16 sW[192*CR_];
  int bt = blockIdx.x;
  int n0 = blockIdx.y * 128;
  tdm_load_2d_f16((unsigned)(size_t)sW, wh, /*d0*/CR_, /*d1*/192,
                  /*tile0*/CR_, /*tile1*/192, /*stride0*/CR_);
  __builtin_amdgcn_s_wait_tensorcnt(0);
  __syncthreads();
  int lane = threadIdx.x & 31;
  int ncol = lane & 15, mhi = (lane & 16) ? 8 : 0;
  for (int rt = 0; rt < 8; ++rt) {
    const float* A = xa + ((size_t)bt*N_ + n0 + rt*16)*CR_;
    v16h a0 = ld_frag_f(A,      CR_, 1);
    v16h a1 = ld_frag_f(A + 32, CR_, 1);
    for (int ct = 0; ct < 12; ++ct) {
      const _Float16* Bp = sW + (size_t)(ct*16)*CR_;   // element (k=c, n=o) = w[o*64+c]
      v16h b0 = ld_frag_h(Bp,      CR_, 1);
      v16h b1 = ld_frag_h(Bp + 32, CR_, 1);
      v8f acc = {};
      acc = WMMA(a0, b0, acc);
      acc = WMMA(a1, b1, acc);
      int o = ct*16 + ncol;
      float bv = bias[o];
      _Float16* dst; int c;
      if (ct < 4)      { dst = qh; c = o; }
      else if (ct < 8) { dst = kh; c = o - 64; }
      else             { dst = vh; c = o - 128; }
#pragma unroll
      for (int r = 0; r < 8; ++r)
        dst[((size_t)bt*N_ + n0 + rt*16 + r + mhi)*CR_ + c] = (_Float16)(acc[r] + bv);
    }
  }
}

// colsums of softmax(QK^T/8) rows at t = TP-1, accumulated per column (atomic)
__global__ __launch_bounds__(32)
void k_colsum(const _Float16* qh, const _Float16* kh, float* colsum) {
  int bn = blockIdx.x; int b = bn / N_; int n = bn % N_;
  int lane = threadIdx.x;
  size_t base = (((size_t)b*TP_ + (TP_-1))*N_)*CR_;
  const _Float16* q = qh + base + (size_t)n*CR_;
  float sc[16];
#pragma unroll 1
  for (int i = 0; i < 16; ++i) {
    int m = lane + 32*i;
    const _Float16* kp = kh + base + (size_t)m*CR_;
    float s = 0.f;
    for (int c = 0; c < CR_; ++c) s += (float)q[c]*(float)kp[c];
    sc[i] = s * 0.125f;
  }
  float mx = -1e30f;
  for (int i = 0; i < 16; ++i) mx = fmaxf(mx, sc[i]);
  for (int off = 16; off; off >>= 1) mx = fmaxf(mx, __shfl_xor(mx, off, 32));
  float sum = 0.f;
  for (int i = 0; i < 16; ++i) { sc[i] = __expf(sc[i]-mx); sum += sc[i]; }
  for (int off = 16; off; off >>= 1) sum += __shfl_xor(sum, off, 32);
  float inv = 1.f/sum;
  for (int i = 0; i < 16; ++i) atomicAdd(&colsum[b*N_ + lane + 32*i], sc[i]*inv);
}

// y[b] = index of 3rd-smallest colsum (stable lowest-index tie break)
__global__ void k_pick(const float* colsum, int* ybuf) {
  int b = blockIdx.x;
  if (threadIdx.x != 0) return;
  int e0 = -1, e1 = -1, yi = 0;
  for (int pass = 0; pass < 3; ++pass) {
    float best = 3.4e38f; int bi = 0;
    for (int m = 0; m < N_; ++m) {
      if (m == e0 || m == e1) continue;
      float v = colsum[b*N_ + m];
      if (v < best) { best = v; bi = m; }
    }
    if (pass == 0) e0 = bi; else if (pass == 1) e1 = bi; else yi = bi;
  }
  ybuf[b] = yi;
}

// mutate v "front" rows at y with uint8-quantized neighbor mean
__global__ void k_mix(_Float16* vh, const int* ybuf) {
  int b = blockIdx.x / TIN_; int t = blockIdx.x % TIN_;
  int c = threadIdx.x;                        // 64 threads
  int y = ybuf[b];
  float s = 0.f; int cnt = 0;
  for (int d = -1; d <= 1; ++d) {
    int col = y + d;
    if (col >= 0 && col < N_) { s += (float)vh[(((size_t)b*TP_ + t)*N_ + col)*CR_ + c]; ++cnt; }
  }
  float mixed = s / (float)cnt;
  mixed = (float)(unsigned char)(int)mixed;   // astype(uint8).astype(f32)
  vh[(((size_t)b*TP_ + t)*N_ + y)*CR_ + c] = (_Float16)mixed;
}

// fused attention: S = QK^T/8 (WMMA) -> softmax in LDS -> vi = P V (WMMA)
#define NPAD 520
__global__ __launch_bounds__(32)
void k_attn(const _Float16* qh, const _Float16* kh, const _Float16* vh, float* vi) {
  __shared__ float    sS[16*NPAD];
  __shared__ _Float16 sP[16*NPAD];
  int bt = blockIdx.x;
  int n0 = blockIdx.y * 16;
  int lane = threadIdx.x;
  int ncol = lane & 15, mhi = (lane & 16) ? 8 : 0;
  const _Float16* Qp = qh + ((size_t)bt*N_ + n0)*CR_;
  v16h a0 = ld_frag_h(Qp,      CR_, 1);
  v16h a1 = ld_frag_h(Qp + 32, CR_, 1);
  for (int kt = 0; kt < N_/16; ++kt) {
    const _Float16* Kp = kh + ((size_t)bt*N_ + kt*16)*CR_;   // B(k=c, n=key)
    v16h b0 = ld_frag_h(Kp,      CR_, 1);
    v16h b1 = ld_frag_h(Kp + 32, CR_, 1);
    v8f acc = {};
    acc = WMMA(a0, b0, acc);
    acc = WMMA(a1, b1, acc);
#pragma unroll
    for (int r = 0; r < 8; ++r) sS[(r+mhi)*NPAD + kt*16 + ncol] = acc[r]*0.125f;
  }
  __syncthreads();
  for (int r = 0; r < 16; ++r) {
    float mx = -1e30f;
    for (int c = lane; c < N_; c += 32) mx = fmaxf(mx, sS[r*NPAD + c]);
    for (int off = 16; off; off >>= 1) mx = fmaxf(mx, __shfl_xor(mx, off, 32));
    float sum = 0.f;
    for (int c = lane; c < N_; c += 32) { float e = __expf(sS[r*NPAD+c]-mx); sS[r*NPAD+c] = e; sum += e; }
    for (int off = 16; off; off >>= 1) sum += __shfl_xor(sum, off, 32);
    float inv = 1.f/sum;
    for (int c = lane; c < N_; c += 32) sP[r*NPAD + c] = (_Float16)(sS[r*NPAD+c]*inv);
  }
  __syncthreads();
  v8f acc[4] = {};
  for (int kc = 0; kc < N_/32; ++kc) {
    v16h ap = ld_frag_h(sP + kc*32, NPAD, 1);
    const _Float16* Vp = vh + ((size_t)bt*N_ + kc*32)*CR_;   // B(k=key, n=ch)
#pragma unroll
    for (int f = 0; f < 4; ++f) {
      v16h bb = ld_frag_h(Vp + f*16, 1, CR_);
      acc[f] = WMMA(ap, bb, acc[f]);
    }
  }
#pragma unroll
  for (int f = 0; f < 4; ++f)
#pragma unroll
    for (int r = 0; r < 8; ++r)
      vi[((size_t)bt*N_ + n0 + r + mhi)*CR_ + f*16 + ncol] = acc[f][r];
}

// FFN (gelu) + residual + layernorm -> conv layout xconv[b][c][n][t].
// Both 64x64 weight panels staged into LDS via two in-flight TDM ops.
__global__ __launch_bounds__(32)
void k_ffn(const float* vi, const float* xa, const _Float16* w1h, const float* b1,
           const _Float16* w2h, const float* b2, const float* lng, const float* lnb,
           float* xconv) {
  __shared__ _Float16 sW1[64*64];
  __shared__ _Float16 sW2[64*64];
  __shared__ _Float16 sH[16*64];
  __shared__ float    sO[16*64];
  tdm_load_2d_f16((unsigned)(size_t)sW1, w1h, CR_, CR_, CR_, CR_, CR_);
  tdm_load_2d_f16((unsigned)(size_t)sW2, w2h, CR_, CR_, CR_, CR_, CR_);
  int bt = blockIdx.x; int b = bt / TP_; int t = bt % TP_;
  int n0 = blockIdx.y * 16;
  int lane = threadIdx.x;
  int ncol = lane & 15, mhi = (lane & 16) ? 8 : 0;
  const float* Ap = vi + ((size_t)bt*N_ + n0)*CR_;
  v16h a0 = ld_frag_f(Ap,      CR_, 1);
  v16h a1 = ld_frag_f(Ap + 32, CR_, 1);
  __builtin_amdgcn_s_wait_tensorcnt(0);
  __syncthreads();
  for (int ct = 0; ct < 4; ++ct) {
    const _Float16* Bp = sW1 + (size_t)(ct*16)*CR_;
    v16h b0 = ld_frag_h(Bp,      CR_, 1);
    v16h b1f = ld_frag_h(Bp + 32, CR_, 1);
    v8f acc = {};
    acc = WMMA(a0, b0, acc);
    acc = WMMA(a1, b1f, acc);
    float bv = b1[ct*16 + ncol];
#pragma unroll
    for (int r = 0; r < 8; ++r) {
      float x = acc[r] + bv;
      sH[(r+mhi)*64 + ct*16 + ncol] = (_Float16)(0.5f*x*(1.f + erff(x*0.70710678f)));
    }
  }
  __syncthreads();
  v16h h0 = ld_frag_h(sH,      64, 1);
  v16h h1 = ld_frag_h(sH + 32, 64, 1);
  for (int ct = 0; ct < 4; ++ct) {
    const _Float16* Bp = sW2 + (size_t)(ct*16)*CR_;
    v16h b0 = ld_frag_h(Bp,      CR_, 1);
    v16h b1f = ld_frag_h(Bp + 32, CR_, 1);
    v8f acc = {};
    acc = WMMA(h0, b0, acc);
    acc = WMMA(h1, b1f, acc);
    float bv = b2[ct*16 + ncol];
#pragma unroll
    for (int r = 0; r < 8; ++r) {
      int m = r + mhi, c = ct*16 + ncol;
      sO[m*64 + c] = acc[r] + bv + xa[((size_t)bt*N_ + n0 + m)*CR_ + c];
    }
  }
  __syncthreads();
  int r = lane & 15, hf = lane >> 4;
  float s = 0.f;
  for (int i = 0; i < 32; ++i) s += sO[r*64 + hf*32 + i];
  s += __shfl_xor(s, 16, 32);
  float mean = s * (1.f/64.f);
  float var = 0.f;
  for (int i = 0; i < 32; ++i) { float d = sO[r*64 + hf*32 + i] - mean; var += d*d; }
  var += __shfl_xor(var, 16, 32);
  float inv = rsqrtf(var*(1.f/64.f) + 1e-5f);
  for (int i = 0; i < 32; ++i) {
    int c = hf*32 + i;
    xconv[(((size_t)b*CR_ + c)*N_ + n0 + r)*TP_ + t] = (sO[r*64+c]-mean)*inv*lng[c] + lnb[c];
  }
}

// skip0: full-time conv of inp
__global__ void k_skip0(const float* inp, const float* wgt, const float* bias, float* skip) {
  int i = blockIdx.x*256 + threadIdx.x;
  if (i >= B_*CS_*N_) return;
  int n = i % N_; int cs = (i/N_) % CS_; int b = i/(N_*CS_);
  float s = bias[cs];
  for (int f = 0; f < FIN_; ++f) {
    const float* xp = inp + (((size_t)b*FIN_ + f)*N_ + n)*TP_;
    const float* wp = wgt + (size_t)(cs*FIN_ + f)*TP_;
    for (int t = 0; t < TP_; ++t) s += xp[t]*wp[t];
  }
  skip[i] = s;
}

// dilated inception: filter/gate WMMA over channel dim per tap, tanh*sigmoid gate
__global__ __launch_bounds__(32)
void k_incept(const _Float16* xh, const _Float16* fw, const _Float16* gw,
              const float* fb0, const float* fb1, const float* fb2, const float* fb3,
              const float* gb0, const float* gb1, const float* gb2, const float* gb3,
              float* gact, int T) {
  int b = blockIdx.x >> 5; int n0 = (blockIdx.x & 31) * 16;
  int tout = blockIdx.y;
  int lane = threadIdx.x;
  int ncol = lane & 15, mhi = (lane & 16) ? 8 : 0;
  const int KK[4] = {2, 3, 6, 7};
  const int WO[4] = {0, 16*64*2, 16*64*5, 16*64*11};
  const float* fbs[4] = {fb0, fb1, fb2, fb3};
  const float* gbs[4] = {gb0, gb1, gb2, gb3};
  for (int j = 0; j < 4; ++j) {
    int kk = KK[j];
    v8f fa = {}, ga = {};
    for (int dt = 0; dt < kk; ++dt) {
      int tin = tout + 7 - kk + dt;
      // A(m=n_local, k=c) = xh[((b*64+c)*N + n0+m)*TP + tin]
      const _Float16* Ap = xh + (((size_t)b*CR_)*N_ + n0)*TP_ + tin;
      v16h a0 = ld_frag_h(Ap,                        TP_, N_*TP_);
      v16h a1 = ld_frag_h(Ap + (size_t)32*N_*TP_,    TP_, N_*TP_);
      // B(k=c, n=oc) = w[(oc*64+c)*kk + dt]
      const _Float16* Fp = fw + WO[j] + dt;
      v16h f0 = ld_frag_h(Fp,           64*kk, kk);
      v16h f1 = ld_frag_h(Fp + 32*kk,   64*kk, kk);
      fa = WMMA(a0, f0, fa);
      fa = WMMA(a1, f1, fa);
      const _Float16* Gp = gw + WO[j] + dt;
      v16h g0 = ld_frag_h(Gp,           64*kk, kk);
      v16h g1 = ld_frag_h(Gp + 32*kk,   64*kk, kk);
      ga = WMMA(a0, g0, ga);
      ga = WMMA(a1, g1, ga);
    }
    float fbv = fbs[j][ncol], gbv = gbs[j][ncol];
#pragma unroll
    for (int r = 0; r < 8; ++r) {
      float f = tanhf(fa[r] + fbv);
      float g = 1.f/(1.f + __expf(-(ga[r] + gbv)));
      int cc = j*16 + ncol;
      gact[(((size_t)b*CC_ + cc)*N_ + n0 + r + mhi)*18 + tout] = f*g;
    }
  }
}

// skip += conv over full remaining time (klen)
__global__ void k_skipadd(const float* src, const float* wgt, const float* bias,
                          float* skip, int klen, int tstride) {
  int i = blockIdx.x*256 + threadIdx.x;
  if (i >= B_*CS_*N_) return;
  int n = i % N_; int cs = (i/N_) % CS_; int b = i/(N_*CS_);
  float s = bias[cs];
  for (int c = 0; c < 64; ++c) {
    const float* xp = src + (((size_t)b*64 + c)*N_ + n)*tstride;
    __builtin_prefetch(xp + (size_t)N_*tstride, 0, 0);
    const float* wp = wgt + (size_t)(cs*64 + c)*klen;
    for (int t = 0; t < klen; ++t) s += xp[t]*wp[t];
  }
  skip[i] += s;
}

// res 1x1 conv + residual (time-shifted by 6)
__global__ void k_res(const float* gact, const float* wgt, const float* bias,
                      const float* xold, float* xnew, int Tnew) {
  int total = B_*CR_*N_*Tnew;
  int i = blockIdx.x*256 + threadIdx.x;
  if (i >= total) return;
  int t = i % Tnew; int n = (i/Tnew) % N_; int c = (i/(Tnew*N_)) % CR_; int b = i/(Tnew*N_*CR_);
  float s = bias[c];
  const float* gp = gact + (((size_t)b*CC_)*N_ + n)*18 + t;
  for (int cc = 0; cc < CC_; ++cc) s += gp[(size_t)cc*N_*18] * wgt[c*CC_ + cc];
  size_t o = (((size_t)b*CR_ + c)*N_ + n)*TP_;
  xnew[o + t] = s + xold[o + t + 6];
}

__global__ void k_skiprelu(const float* skip, float* xemb, _Float16* sh, int n) {
  int i = blockIdx.x*256 + threadIdx.x;
  if (i < n) { float v = fmaxf(skip[i], 0.f); xemb[i] = v; sh[i] = (_Float16)v; }
}

// end1: (512x128)x(128x256) WMMA GEMM per batch, relu
__global__ __launch_bounds__(32)
void k_end1(const _Float16* sh, const _Float16* wh, const float* bias, float* e1) {
  int b = blockIdx.x; int n0 = blockIdx.y * 16;
  const _Float16* Ap = sh + ((size_t)b*CS_)*N_ + n0;   // A(m=n_local, k=cs): strideM=1, strideK=N
  v16h a[4];
#pragma unroll
  for (int kc = 0; kc < 4; ++kc) a[kc] = ld_frag_h(Ap + (size_t)kc*32*N_, 1, N_);
  int lane = threadIdx.x;
  int ncol = lane & 15, mhi = (lane & 16) ? 8 : 0;
  for (int ct = 0; ct < 16; ++ct) {
    v8f acc = {};
#pragma unroll
    for (int kc = 0; kc < 4; ++kc) {
      const _Float16* Bp = wh + (size_t)(ct*16)*CS_ + kc*32;  // B(k=cs,n=ce)=wh[ce*128+cs]
      v16h bb = ld_frag_h(Bp, CS_, 1);
      acc = WMMA(a[kc], bb, acc);
    }
    int ce = ct*16 + ncol;
    float bv = bias[ce];
#pragma unroll
    for (int r = 0; r < 8; ++r)
      e1[((size_t)b*CE_ + ce)*N_ + n0 + r + mhi] = fmaxf(acc[r] + bv, 0.f);
  }
}

__global__ void k_end2(const float* e1, const float* wgt, const float* bias, float* out) {
  int i = blockIdx.x*256 + threadIdx.x;
  if (i >= B_*TIN_*N_) return;
  int n = i % N_; int to = (i/N_) % TIN_; int b = i/(N_*TIN_);
  float s = bias[to];
  const float* ep = e1 + ((size_t)b*CE_)*N_ + n;
  for (int ce = 0; ce < CE_; ++ce) s += ep[(size_t)ce*N_]*wgt[to*CE_ + ce];
  out[i] = s;
}

extern "C" void kernel_launch(void* const* d_in, const int* in_sizes, int n_in,
                              void* d_out, int out_size, void* d_ws, size_t ws_size,
                              hipStream_t stream) {
  (void)in_sizes; (void)n_in; (void)out_size; (void)ws_size;
  const float* src    = (const float*)d_in[0];
  const float* plus_w = (const float*)d_in[1];
  const float* plus_b = (const float*)d_in[2];
  const float* ext_w  = (const float*)d_in[3];
  const float* ext_b  = (const float*)d_in[4];
  const float* xff_w  = (const float*)d_in[5];
  const float* xff_b  = (const float*)d_in[6];
  const float* ff1_w  = (const float*)d_in[7];
  const float* ff1_b  = (const float*)d_in[8];
  const float* ff2_w  = (const float*)d_in[9];
  const float* ff2_b  = (const float*)d_in[10];
  const float* ln_g   = (const float*)d_in[11];
  const float* ln_b   = (const float*)d_in[12];
  const float* sk0_w  = (const float*)d_in[13];
  const float* sk0_b  = (const float*)d_in[14];
  const float* skE_w  = (const float*)d_in[15];
  const float* skE_b  = (const float*)d_in[16];
  // filter_w: 17..28, filter_b: 29..40, gate_w: 41..52, gate_b: 53..64
  // skip_w: 65..67, skip_b: 68..70, res_w: 71..73, res_b: 74..76
  const float* e1_w = (const float*)d_in[77];
  const float* e1_b = (const float*)d_in[78];
  const float* e2_w = (const float*)d_in[79];
  const float* e2_b = (const float*)d_in[80];
  float* out = (float*)d_out;

  const size_t SZ_XA = (size_t)B_*TP_*N_*CR_;           // 12,582,912 (== B*64*N*TP)
  char* w = (char*)d_ws;
  size_t off = 0;
  auto alloc = [&](size_t bytes) { size_t o = off; off += (bytes + 255) & ~(size_t)255; return o; };
  float*    inp    = (float*)   (w + alloc((size_t)B_*FIN_*N_*TP_*4));
  float*    xa     = (float*)   (w + alloc(SZ_XA*4));     // reused as gact later
  _Float16* qh     = (_Float16*)(w + alloc(SZ_XA*2));     // reused as xh later
  _Float16* kh     = (_Float16*)(w + alloc(SZ_XA*2));     // reused as e1 later
  _Float16* vh     = (_Float16*)(w + alloc(SZ_XA*2));     // reused as sh later
  float*    vi     = (float*)   (w + alloc(SZ_XA*4));     // reused as x ping buffer
  float*    xf0    = (float*)   (w + alloc(SZ_XA*4));
  float*    skip   = (float*)   (w + alloc((size_t)B_*CS_*N_*4));
  float*    colsum = (float*)   (w + alloc((size_t)B_*N_*4));
  int*      ybuf   = (int*)     (w + alloc(256));
  _Float16* whq    = (_Float16*)(w + alloc(192*64*2));
  _Float16* wh1    = (_Float16*)(w + alloc(64*64*2));
  _Float16* wh2    = (_Float16*)(w + alloc(64*64*2));
  _Float16* whe1   = (_Float16*)(w + alloc(256*128*2));
  _Float16* fwp    = (_Float16*)(w + alloc((size_t)3*18432*2));
  _Float16* gwp    = (_Float16*)(w + alloc((size_t)3*18432*2));
  float*    gact   = xa;
  _Float16* xh     = qh;
  float*    e1     = (float*)kh;
  _Float16* sh     = vh;

  // weight conversions (f32 -> f16)
  k_f2h<<<(192*64+255)/256, 256, 0, stream>>>(xff_w, whq, 192*64);
  k_f2h<<<(64*64+255)/256,  256, 0, stream>>>(ff1_w, wh1, 64*64);
  k_f2h<<<(64*64+255)/256,  256, 0, stream>>>(ff2_w, wh2, 64*64);
  k_f2h<<<(256*128+255)/256,256, 0, stream>>>(e1_w, whe1, 256*128);
  const int KK[4] = {2, 3, 6, 7};
  const int WO[4] = {0, 16*64*2, 16*64*5, 16*64*11};
  for (int i = 0; i < 3; ++i)
    for (int j = 0; j < 4; ++j) {
      int cnt = 16*64*KK[j];
      k_f2h<<<(cnt+255)/256, 256, 0, stream>>>((const float*)d_in[17 + i*4 + j], fwp + i*18432 + WO[j], cnt);
      k_f2h<<<(cnt+255)/256, 256, 0, stream>>>((const float*)d_in[41 + i*4 + j], gwp + i*18432 + WO[j], cnt);
    }

  // front end
  {
    int n = B_*FIN_*N_*TP_;
    k_embed<<<(n+255)/256, 256, 0, stream>>>(src, plus_w, plus_b, inp);
  }
  {
    int n = (int)SZ_XA;
    k_xa<<<(n+255)/256, 256, 0, stream>>>(inp, ext_w, ext_b, xa);
  }
  dim3 gqkv(B_*TP_, N_/128);
  k_qkv<<<gqkv, 32, 0, stream>>>(xa, whq, xff_b, qh, kh, vh);

  // data-dependent v mutation
  k_zerof<<<(B_*N_+255)/256, 256, 0, stream>>>(colsum, B_*N_);
  k_colsum<<<B_*N_, 32, 0, stream>>>(qh, kh, colsum);
  k_pick<<<B_, 32, 0, stream>>>(colsum, ybuf);
  k_mix<<<B_*TIN_, 64, 0, stream>>>(vh, ybuf);

  // attention + FFN/LN
  dim3 gbt(B_*TP_, N_/16);
  k_attn<<<gbt, 32, 0, stream>>>(qh, kh, vh, vi);
  k_ffn<<<gbt, 32, 0, stream>>>(vi, xa, wh1, ff1_b, wh2, ff2_b, ln_g, ln_b, xf0);

  // skip0
  k_skip0<<<(B_*CS_*N_+255)/256, 256, 0, stream>>>(inp, sk0_w, sk0_b, skip);

  // TCN layers
  const int T_arr[3] = {24, 18, 12};
  float* xc = xf0; float* xn = vi;
  for (int i = 0; i < 3; ++i) {
    int T = T_arr[i], Tnew = T - 6;
    k_f2h<<<((int)SZ_XA + 255)/256, 256, 0, stream>>>(xc, xh, (int)SZ_XA);
    dim3 gi(B_*32, Tnew);
    k_incept<<<gi, 32, 0, stream>>>(xh, fwp + i*18432, gwp + i*18432,
        (const float*)d_in[29+i*4+0], (const float*)d_in[29+i*4+1],
        (const float*)d_in[29+i*4+2], (const float*)d_in[29+i*4+3],
        (const float*)d_in[53+i*4+0], (const float*)d_in[53+i*4+1],
        (const float*)d_in[53+i*4+2], (const float*)d_in[53+i*4+3],
        gact, T);
    k_skipadd<<<(B_*CS_*N_+255)/256, 256, 0, stream>>>(gact, (const float*)d_in[65+i],
        (const float*)d_in[68+i], skip, Tnew, 18);
    int total = B_*CR_*N_*Tnew;
    k_res<<<(total+255)/256, 256, 0, stream>>>(gact, (const float*)d_in[71+i],
        (const float*)d_in[74+i], xc, xn, Tnew);
    float* tmp = xc; xc = xn; xn = tmp;
  }

  // skipE + outputs
  k_skipadd<<<(B_*CS_*N_+255)/256, 256, 0, stream>>>(xc, skE_w, skE_b, skip, 6, TP_);
  k_skiprelu<<<(B_*CS_*N_+255)/256, 256, 0, stream>>>(skip, out + (size_t)B_*TIN_*N_, sh, B_*CS_*N_);
  dim3 ge1(B_, N_/16);
  k_end1<<<ge1, 32, 0, stream>>>(sh, whe1, e1_b, e1);
  k_end2<<<(B_*TIN_*N_+255)/256, 256, 0, stream>>>(e1, e2_w, e2_b, out);
}
